// MultiHeadAttentionWithKVCache_43181601194828
// MI455X (gfx1250) — matrix operand order repaired and checked by reading, not compiled, not run on hardware
//
#include <hip/hip_runtime.h>

#define HIDDEN 2048
#define NHEAD  16
#define DHEAD  128
#define BSZ    4
#define SEQ    2048
#define NEGMASK (-1e9f)
#define NEGINF  (-1e30f)

typedef __attribute__((ext_vector_type(16))) __bf16 v16bf;
typedef __attribute__((ext_vector_type(8)))  float  v8f;

union Frag {
    v16bf v;
    unsigned int u[8];
    uint4 q[2];
};

__device__ __forceinline__ unsigned short f2bf(float f) {
    unsigned int u = __float_as_uint(f);
    u += 0x7fffu + ((u >> 16) & 1u);       // round-to-nearest-even
    return (unsigned short)(u >> 16);
}

// ---------------------------------------------------------------------------
// f32 -> bf16, 4 elements / thread
__global__ void cvt_bf16_vec(const float* __restrict__ in,
                             unsigned short* __restrict__ out) {
    int i = (blockIdx.x * blockDim.x + threadIdx.x) * 4;
    float4 f = *(const float4*)(in + i);
    ushort4 r;
    r.x = f2bf(f.x); r.y = f2bf(f.y); r.z = f2bf(f.z); r.w = f2bf(f.w);
    *(ushort4*)(out + i) = r;
}

// W [K=2048][N=2048] f32  ->  Wt [N][K] bf16  (transpose so B-frags load contiguously)
__global__ void cvt_transpose_bf16(const float* __restrict__ in,
                                   unsigned short* __restrict__ out) {
    int idx = blockIdx.x * blockDim.x + threadIdx.x;
    int k = idx >> 11, n = idx & 2047;
    out[((size_t)n << 11) | k] = f2bf(in[idx]);
}

// ---------------------------------------------------------------------------
// C = A(bf16 [M][2048]) * Wt(bf16 [N][2048], i.e. W^T) + bias
// MODE 0: store bf16 as [b][h][s][d]   (Q, K)
// MODE 1: store bf16 as [b][h][d][s]   (V transposed, packed 16B stores)
// MODE 2: store f32  as [r][c]         (final projection)
// Block: 256 thr = 8 waves (2x4), block tile 128x128, wave tile 64x32.
// Double-buffered fragment loads (manual 2x unroll, no runtime-indexed arrays).
template <int MODE>
__global__ __launch_bounds__(256, 1)
void gemm_bf16(const unsigned short* __restrict__ A,
               const unsigned short* __restrict__ WT,
               const float* __restrict__ bias,
               void* __restrict__ Out) {
    const int K = 2048;
    int lane = threadIdx.x & 31;
    int w    = threadIdx.x >> 5;
    int lg = lane >> 4, ll = lane & 15;
    int m0 = blockIdx.y * 128 + (w & 1) * 64;
    int n0 = blockIdx.x * 128 + (w >> 1) * 32;

    const unsigned short* Abase = A  + (size_t)(m0 + ll) * K + lg * 8;
    const unsigned short* Bbase = WT + (size_t)(n0 + ll) * K + lg * 8;

    v8f acc[4][2];
#pragma unroll
    for (int mt = 0; mt < 4; ++mt)
#pragma unroll
        for (int nt = 0; nt < 2; ++nt)
#pragma unroll
            for (int i = 0; i < 8; ++i) acc[mt][nt][i] = 0.f;

    Frag a0[4], b0[2], a1[4], b1[2];

    auto loadA = [&](Frag (&a)[4], int k) {
#pragma unroll
        for (int mt = 0; mt < 4; ++mt) {
            const unsigned short* p = Abase + (size_t)mt * 16 * K + k;
            a[mt].q[0] = *(const uint4*)p;          // K = k + lg*8 .. +7
            a[mt].q[1] = *(const uint4*)(p + 16);   // K = k+16+lg*8 .. +7
        }
    };
    auto loadB = [&](Frag (&b)[2], int k) {
#pragma unroll
        for (int nt = 0; nt < 2; ++nt) {
            const unsigned short* p = Bbase + (size_t)nt * 16 * K + k;
            b[nt].q[0] = *(const uint4*)p;          // K = k + lg*8 .. +7
            b[nt].q[1] = *(const uint4*)(p + 16);
        }
    };
    auto compute = [&](Frag (&a)[4], Frag (&b)[2]) {
#pragma unroll
        for (int mt = 0; mt < 4; ++mt)
#pragma unroll
            for (int nt = 0; nt < 2; ++nt)
                acc[mt][nt] = __builtin_amdgcn_wmma_f32_16x16x32_bf16(
                    false, a[mt].v, false, b[nt].v, (short)0, acc[mt][nt], false, false);
    };

    loadA(a0, 0); loadB(b0, 0);
    for (int k = 0; k < K; k += 64) {
        loadA(a1, k + 32); loadB(b1, k + 32);   // prefetch next chunk
        compute(a0, b0);
        if (k + 64 < K) { loadA(a0, k + 64); loadB(b0, k + 64); }
        compute(a1, b1);
    }

#pragma unroll
    for (int mt = 0; mt < 4; ++mt) {
#pragma unroll
        for (int nt = 0; nt < 2; ++nt) {
            int cg = n0 + nt * 16 + ll;           // lane's column
            float bv = bias[cg];
            if (MODE == 2) {
                float* out = (float*)Out;
#pragma unroll
                for (int v = 0; v < 8; ++v) {
                    int rg = m0 + mt * 16 + lg * 8 + v;
                    out[(size_t)rg * 2048 + cg] = acc[mt][nt][v] + bv;
                }
            } else if (MODE == 0) {
                unsigned short* out = (unsigned short*)Out;
                int h = cg >> 7, d = cg & 127;
#pragma unroll
                for (int v = 0; v < 8; ++v) {
                    int rg = m0 + mt * 16 + lg * 8 + v;
                    int bb = rg >> 11, s = rg & 2047;
                    out[((((size_t)bb * NHEAD + h) * SEQ + s) << 7) + d] =
                        f2bf(acc[mt][nt][v] + bv);
                }
            } else {  // MODE 1: V^T [b][h][d][s], rows v are 8 consecutive s
                unsigned short* out = (unsigned short*)Out;
                int h = cg >> 7, d = cg & 127;
                int r0 = m0 + mt * 16 + lg * 8;
                int bb = r0 >> 11, s0 = r0 & 2047;
                uint4 val;
                val.x = (unsigned)f2bf(acc[mt][nt][0] + bv) | ((unsigned)f2bf(acc[mt][nt][1] + bv) << 16);
                val.y = (unsigned)f2bf(acc[mt][nt][2] + bv) | ((unsigned)f2bf(acc[mt][nt][3] + bv) << 16);
                val.z = (unsigned)f2bf(acc[mt][nt][4] + bv) | ((unsigned)f2bf(acc[mt][nt][5] + bv) << 16);
                val.w = (unsigned)f2bf(acc[mt][nt][6] + bv) | ((unsigned)f2bf(acc[mt][nt][7] + bv) << 16);
                *(uint4*)(out + ((((size_t)bb * NHEAD + h) * DHEAD + d) << 11) + s0) = val;
            }
        }
    }
}

// ---------------------------------------------------------------------------
// Flash attention: each wave owns 16 query rows of one (b,h); online softmax.
// Q,K: [b][h][s][d] bf16, Vt: [b][h][d][s] bf16, O: [b][s][h*128+d] bf16
__global__ __launch_bounds__(256, 1)
void flash_attn(const unsigned short* __restrict__ Q,
                const unsigned short* __restrict__ Kc,
                const unsigned short* __restrict__ Vt,
                unsigned short* __restrict__ O) {
    __shared__ unsigned short pstage[8][16 * 32];
    const float scale = 0.08838834764831845f;  // 1/sqrt(128)
    int lane = threadIdx.x & 31;
    int w = threadIdx.x >> 5;
    int lg = lane >> 4, ll = lane & 15;
    int b = blockIdx.z, h = blockIdx.y;
    int qbase = blockIdx.x * 128 + w * 16;

    const unsigned short* qp = Q  + ((((size_t)b * NHEAD + h) * SEQ) << 7);
    const unsigned short* kp = Kc + ((((size_t)b * NHEAD + h) * SEQ) << 7);
    const unsigned short* vp = Vt + ((((size_t)b * NHEAD + h) * DHEAD) << 11);

    // persistent Q fragments: 4 chunks of d (WMMA-K = d)
    Frag qf[4];
#pragma unroll
    for (int c = 0; c < 4; ++c) {
        const unsigned short* p = qp + ((size_t)(qbase + ll) << 7) + c * 32 + lg * 8;
        qf[c].q[0] = *(const uint4*)p;
        qf[c].q[1] = *(const uint4*)(p + 16);
    }

    v8f o[8];
#pragma unroll
    for (int t = 0; t < 8; ++t)
#pragma unroll
        for (int i = 0; i < 8; ++i) o[t][i] = 0.f;
    float mrow[8], lrow[8];
#pragma unroll
    for (int v = 0; v < 8; ++v) { mrow[v] = NEGINF; lrow[v] = 0.f; }

    int nkv = (qbase + 16 + 31) >> 5;  // causal: only kv tiles intersecting lower triangle
    for (int kt = 0; kt < nkv; ++kt) {
        int kvbase = kt * 32;
        // ---- batch-load the 8 K fragments of this kv tile ----
        Frag kf[2][4];
#pragma unroll
        for (int nt = 0; nt < 2; ++nt)
#pragma unroll
            for (int c = 0; c < 4; ++c) {
                const unsigned short* p =
                    kp + ((size_t)(kvbase + nt * 16 + ll) << 7) + c * 32 + lg * 8;
                kf[nt][c].q[0] = *(const uint4*)p;
                kf[nt][c].q[1] = *(const uint4*)(p + 16);
            }
        // ---- S = Q K^T (two 16x16 tiles, back-to-back WMMAs) ----
        v8f s[2];
#pragma unroll
        for (int nt = 0; nt < 2; ++nt)
#pragma unroll
            for (int i = 0; i < 8; ++i) s[nt][i] = 0.f;
#pragma unroll
        for (int nt = 0; nt < 2; ++nt)
#pragma unroll
            for (int c = 0; c < 4; ++c)
                s[nt] = __builtin_amdgcn_wmma_f32_16x16x32_bf16(
                    false, qf[c].v, false, kf[nt][c].v, (short)0, s[nt], false, false);
        // ---- issue V fragment loads now; they complete under the softmax VALU ----
        Frag vf[8];
#pragma unroll
        for (int t = 0; t < 8; ++t) {
            const unsigned short* p = vp + ((size_t)(t * 16 + ll) << 11) + kvbase + lg * 8;
            vf[t].q[0] = *(const uint4*)p;
            vf[t].q[1] = *(const uint4*)(p + 16);
        }
        // ---- scale + analytic causal mask + row max ----
        float smax[8];
#pragma unroll
        for (int v = 0; v < 8; ++v) {
            int qrow = qbase + lg * 8 + v;
            float s0 = s[0][v] * scale + ((kvbase + ll)      > qrow ? NEGMASK : 0.f);
            float s1 = s[1][v] * scale + ((kvbase + 16 + ll) > qrow ? NEGMASK : 0.f);
            s[0][v] = s0; s[1][v] = s1;
            smax[v] = fmaxf(s0, s1);
        }
#pragma unroll
        for (int v = 0; v < 8; ++v)
#pragma unroll
            for (int off = 1; off < 16; off <<= 1)
                smax[v] = fmaxf(smax[v], __shfl_xor(smax[v], off, 32));
        // ---- online softmax update ----
        float corr[8], rs[8];
#pragma unroll
        for (int v = 0; v < 8; ++v) {
            float mnew = fmaxf(mrow[v], smax[v]);
            corr[v] = __expf(mrow[v] - mnew);
            mrow[v] = mnew;
            float p0 = __expf(s[0][v] - mnew);
            float p1 = __expf(s[1][v] - mnew);
            s[0][v] = p0; s[1][v] = p1;
            rs[v] = p0 + p1;
        }
#pragma unroll
        for (int v = 0; v < 8; ++v) {
#pragma unroll
            for (int off = 1; off < 16; off <<= 1)
                rs[v] += __shfl_xor(rs[v], off, 32);
            lrow[v] = lrow[v] * corr[v] + rs[v];
        }
#pragma unroll
        for (int t = 0; t < 8; ++t)
#pragma unroll
            for (int v = 0; v < 8; ++v) o[t][v] *= corr[v];
        // ---- stage P (C-layout) into LDS, read back in A-layout ----
        unsigned short* ps = pstage[w];
#pragma unroll
        for (int nt = 0; nt < 2; ++nt)
#pragma unroll
            for (int v = 0; v < 8; ++v)
                ps[(lg * 8 + v) * 32 + nt * 16 + ll] = f2bf(s[nt][v]);
        Frag pf;  // same-wave DS ops are in-order: no barrier needed
        {
            const unsigned short* p = ps + ll * 32 + lg * 8;
            pf.q[0] = *(const uint4*)p;
            pf.q[1] = *(const uint4*)(p + 16);
        }
        // ---- O += P * V (WMMA-K = kv dim; V loads already in flight) ----
#pragma unroll
        for (int t = 0; t < 8; ++t)
            o[t] = __builtin_amdgcn_wmma_f32_16x16x32_bf16(
                false, pf.v, false, vf[t].v, (short)0, o[t], false, false);
    }
    // ---- normalize and store ----
#pragma unroll
    for (int v = 0; v < 8; ++v) lrow[v] = 1.f / lrow[v];
#pragma unroll
    for (int t = 0; t < 8; ++t) {
        int d = t * 16 + ll;
#pragma unroll
        for (int v = 0; v < 8; ++v) {
            int srow = qbase + lg * 8 + v;
            O[(((size_t)b * SEQ + srow) << 11) + h * DHEAD + d] = f2bf(o[t][v] * lrow[v]);
        }
    }
}

// ---------------------------------------------------------------------------
extern "C" void kernel_launch(void* const* d_in, const int* in_sizes, int n_in,
                              void* d_out, int out_size, void* d_ws, size_t ws_size,
                              hipStream_t stream) {
    const float* x  = (const float*)d_in[0];
    // d_in[1] = causal_mask: applied analytically (exact causal), never read
    const float* Wq = (const float*)d_in[2];
    const float* bq = (const float*)d_in[3];
    const float* Wk = (const float*)d_in[4];
    const float* bk = (const float*)d_in[5];
    const float* Wv = (const float*)d_in[6];
    const float* bv = (const float*)d_in[7];
    const float* Wo = (const float*)d_in[8];
    const float* bo = (const float*)d_in[9];
    float* out = (float*)d_out;

    char* ws = (char*)d_ws;
    const size_t XB  = 0;                       // 32 MiB  x as bf16
    const size_t WQT = XB  + (33554432);        // 8 MiB each, W^T bf16
    const size_t WKT = WQT + (8388608);
    const size_t WVT = WKT + (8388608);
    const size_t WOT = WVT + (8388608);
    const size_t QB  = WOT + (8388608);         // 32 MiB  Q [b][h][s][d]
    const size_t KB  = QB  + (33554432);        // 32 MiB  K [b][h][s][d]
    const size_t VT  = KB  + (33554432);        // 32 MiB  V^T [b][h][d][s]
    const size_t AB  = VT  + (33554432);        // 32 MiB  attn out [r][c]
    unsigned short* xb  = (unsigned short*)(ws + XB);
    unsigned short* wqt = (unsigned short*)(ws + WQT);
    unsigned short* wkt = (unsigned short*)(ws + WKT);
    unsigned short* wvt = (unsigned short*)(ws + WVT);
    unsigned short* wot = (unsigned short*)(ws + WOT);
    unsigned short* qb  = (unsigned short*)(ws + QB);
    unsigned short* kb  = (unsigned short*)(ws + KB);
    unsigned short* vt  = (unsigned short*)(ws + VT);
    unsigned short* ab  = (unsigned short*)(ws + AB);

    const int M = BSZ * SEQ;          // 8192
    const int NX = M * HIDDEN;        // 16,777,216

    cvt_bf16_vec<<<NX / 4 / 256, 256, 0, stream>>>(x, xb);
    const int NW = HIDDEN * HIDDEN;   // 4,194,304
    cvt_transpose_bf16<<<NW / 256, 256, 0, stream>>>(Wq, wqt);
    cvt_transpose_bf16<<<NW / 256, 256, 0, stream>>>(Wk, wkt);
    cvt_transpose_bf16<<<NW / 256, 256, 0, stream>>>(Wv, wvt);
    cvt_transpose_bf16<<<NW / 256, 256, 0, stream>>>(Wo, wot);

    dim3 ggrid(HIDDEN / 128, M / 128);            // (16, 64)
    gemm_bf16<0><<<ggrid, 256, 0, stream>>>(xb, wqt, bq, qb);
    gemm_bf16<0><<<ggrid, 256, 0, stream>>>(xb, wkt, bk, kb);
    gemm_bf16<1><<<ggrid, 256, 0, stream>>>(xb, wvt, bv, vt);

    flash_attn<<<dim3(SEQ / 128, NHEAD, BSZ), 256, 0, stream>>>(qb, kb, vt, ab);

    gemm_bf16<2><<<ggrid, 256, 0, stream>>>(ab, wot, bo, out);
}
